// MultiHeadAttention_86096914416219
// MI455X (gfx1250) — matrix operand order repaired and checked by reading, not compiled, not run on hardware
//
#include <hip/hip_runtime.h>
#include <hip/hip_bf16.h>

typedef __bf16 bf16_t;
typedef __attribute__((ext_vector_type(16))) __bf16 v16bf;
typedef __attribute__((ext_vector_type(8)))  __bf16 v8bf;
typedef __attribute__((ext_vector_type(8)))  float  v8f;

#define S_LEN 2048
#define D_DIM 1024
#define H_NUM 16
#define HD_DIM 64
#define B_NUM 2
#define BS_TOK (B_NUM * S_LEN)   // 4096 tokens

// ---------------------------------------------------------------------------
// WMMA helper: D = A(16x32 bf16) * B(32x16 bf16) + C(16x16 f32)
// ---------------------------------------------------------------------------
__device__ __forceinline__ v8f wmma_bf(v16bf a, v16bf b, v8f c) {
  return __builtin_amdgcn_wmma_f32_16x16x32_bf16(
      /*neg_a=*/false, a, /*neg_b=*/false, b,
      /*c_mod=*/(short)0, c, /*reuse_a=*/false, /*reuse_b=*/false);
}

// A-fragment for a 16x32 tile of row-major A[M][ld]:
//   lanes 0-15 : row m0+lo, K = k0+0..15
//   lanes 16-31: row m0+lo, K = k0+16..31
__device__ __forceinline__ v16bf load_a16x32(const bf16_t* __restrict__ A, int ld,
                                             int m0, int k0, int lo, int hi) {
  const bf16_t* p = A + (size_t)(m0 + lo) * ld + k0 + hi * 16;
  v8bf x0 = *(const v8bf*)p;
  v8bf x1 = *(const v8bf*)(p + 8);
  return __builtin_shufflevector(x0, x1, 0,1,2,3,4,5,6,7,8,9,10,11,12,13,14,15);
}

// B-fragment for a 32x16 tile given B^T stored row-major as BT[N][ld] (K contiguous):
//   lanes 0-15 : col n0+lo, K = k0+0..15   (32 contiguous bytes)
//   lanes 16-31: col n0+lo, K = k0+16..31
__device__ __forceinline__ v16bf load_bT32x16(const bf16_t* __restrict__ BT, int ld,
                                              int n0, int k0, int lo, int hi) {
  return *(const v16bf*)(BT + (size_t)(n0 + lo) * ld + k0 + hi * 16);
}

// ---------------------------------------------------------------------------
// Conversion / staging kernels
// ---------------------------------------------------------------------------
__global__ void cvt_x_kernel(const float* __restrict__ src, bf16_t* __restrict__ dst, int n) {
  int i = blockIdx.x * blockDim.x + threadIdx.x;
  int stride = gridDim.x * blockDim.x;
  for (; i < n; i += stride) dst[i] = (bf16_t)src[i];
}

// wt[n][k] = bf16( w[k][n] )   (1024x1024)
__global__ void transpose_w_kernel(const float* __restrict__ w, bf16_t* __restrict__ wt) {
  int idx = blockIdx.x * blockDim.x + threadIdx.x;
  if (idx >= D_DIM * D_DIM) return;
  int k = idx >> 10;
  int n = idx & (D_DIM - 1);
  wt[(size_t)n * D_DIM + k] = (bf16_t)w[idx];
}

// ---------------------------------------------------------------------------
// QKV projection GEMM: [4096,1024] x Wt -> Q,K token-major bf16; V transposed
// per head to vt[B,H,HD,S]. 32x64 output tile per wave (two A-frags share each
// B-frag -> 8 WMMAs per 12 b128 loads per k-step). grid.z selects Q/K/V.
// ---------------------------------------------------------------------------
__global__ void qkv_gemm_kernel(const bf16_t* __restrict__ xb,
                                const bf16_t* __restrict__ wtq,
                                const bf16_t* __restrict__ wtk,
                                const bf16_t* __restrict__ wtv,
                                bf16_t* __restrict__ qb,
                                bf16_t* __restrict__ kbuf,
                                bf16_t* __restrict__ vt) {
  const int lane = threadIdx.x & 31;
  const int wave = threadIdx.x >> 5;
  const int lo = lane & 15, hi = lane >> 4;
  const int m0 = blockIdx.x * 32;
  const int n0 = (blockIdx.y * 8 + wave) * 64;
  const int z  = blockIdx.z;
  const bf16_t* wt = (z == 0) ? wtq : (z == 1) ? wtk : wtv;

  v8f acc[2][4] = {};
  for (int k0 = 0; k0 < D_DIM; k0 += 32) {
    v16bf a0 = load_a16x32(xb, D_DIM, m0,      k0, lo, hi);
    v16bf a1 = load_a16x32(xb, D_DIM, m0 + 16, k0, lo, hi);
#pragma unroll
    for (int t = 0; t < 4; ++t) {
      v16bf bf = load_bT32x16(wt, D_DIM, n0 + t * 16, k0, lo, hi);
      acc[0][t] = wmma_bf(a0, bf, acc[0][t]);
      acc[1][t] = wmma_bf(a1, bf, acc[1][t]);
    }
  }

  if (z < 2) {
    bf16_t* dst = (z == 0) ? qb : kbuf;
#pragma unroll
    for (int i = 0; i < 2; ++i)
#pragma unroll
      for (int t = 0; t < 4; ++t)
#pragma unroll
        for (int r = 0; r < 8; ++r) {
          int m = m0 + i * 16 + r + hi * 8;
          int n = n0 + t * 16 + lo;
          dst[(size_t)m * D_DIM + n] = (bf16_t)acc[i][t][r];
        }
  } else {
#pragma unroll
    for (int i = 0; i < 2; ++i)
#pragma unroll
      for (int t = 0; t < 4; ++t)
#pragma unroll
        for (int r = 0; r < 8; ++r) {
          int m = m0 + i * 16 + r + hi * 8;
          int n = n0 + t * 16 + lo;
          int b = m >> 11, s = m & (S_LEN - 1);
          int h = n >> 6, d = n & 63;
          vt[(size_t)((b * H_NUM + h) * HD_DIM + d) * S_LEN + s] = (bf16_t)acc[i][t][r];
        }
  }
}

// ---------------------------------------------------------------------------
// Flash attention: one wave per (b, h, 32-query tile = two 16-row sub-tiles).
// K/V fragments are loaded once per 32-key block and shared by both sub-tiles.
// ---------------------------------------------------------------------------
__global__ void attn_kernel(const bf16_t* __restrict__ qb,
                            const bf16_t* __restrict__ kbuf,
                            const bf16_t* __restrict__ vt,
                            bf16_t* __restrict__ ctx) {
  __shared__ alignas(16) bf16_t pbuf[2][16 * 32];   // P staging per sub-tile

  const int lane = threadIdx.x & 31;
  const int lo = lane & 15, hi = lane >> 4;
  const int qt = blockIdx.x, h = blockIdx.y, b = blockIdx.z;
  const int q0 = qt * 32;

  // Q A-fragments for both query sub-tiles, head-dim chunks k0 = 0 and 32
  v16bf qa[2][2];
#pragma unroll
  for (int g = 0; g < 2; ++g) {
    const bf16_t* qbase = qb + ((size_t)(b * S_LEN) + q0 + g * 16) * D_DIM + h * HD_DIM;
    qa[g][0] = load_a16x32(qbase, D_DIM, 0, 0,  lo, hi);
    qa[g][1] = load_a16x32(qbase, D_DIM, 0, 32, lo, hi);
  }

  v8f O[2][4] = {};
  float mrow[2][8], lrow[2][8];
#pragma unroll
  for (int g = 0; g < 2; ++g)
#pragma unroll
    for (int r = 0; r < 8; ++r) { mrow[g][r] = -__builtin_inff(); lrow[g][r] = 0.f; }

  const int kblocks = (q0 + 31) / 32 + 1;          // causal: keys <= q0+31
  for (int kb2 = 0; kb2 < kblocks; ++kb2) {
    const int key0 = kb2 * 32;

    // shared K fragments: kf[t][chunk], t = 16-key sub-tile, chunk = head-dim half
    v16bf kf[2][2];
#pragma unroll
    for (int t = 0; t < 2; ++t) {
      const bf16_t* kp = kbuf + ((size_t)(b * S_LEN) + key0 + t * 16 + lo) * D_DIM + h * HD_DIM;
      kf[t][0] = *(const v16bf*)(kp + hi * 16);
      kf[t][1] = *(const v16bf*)(kp + 32 + hi * 16);
    }

    // prefetch next block's K and V lines (global_prefetch_b8)
    if (kb2 + 1 < kblocks) {
      const bf16_t* nkp = kbuf + ((size_t)(b * S_LEN) + key0 + 32 + lo) * D_DIM + h * HD_DIM;
      __builtin_prefetch(nkp, 0, 1);
      const bf16_t* nvp = vt + ((size_t)((b * H_NUM + h) * HD_DIM) + lo) * S_LEN + key0 + 32;
      __builtin_prefetch(nvp, 0, 1);
    }

    // scores + online softmax + P staging, per query sub-tile
#pragma unroll
    for (int g = 0; g < 2; ++g) {
      v8f st[2];
#pragma unroll
      for (int t = 0; t < 2; ++t) {
        v8f a = {};
        a = wmma_bf(qa[g][0], kf[t][0], a);
        a = wmma_bf(qa[g][1], kf[t][1], a);
        st[t] = a;
      }
#pragma unroll
      for (int r = 0; r < 8; ++r) {
        const int qi = q0 + g * 16 + r + hi * 8;
        float s0 = st[0][r] * 0.125f;              // 1/sqrt(64)
        float s1 = st[1][r] * 0.125f;
        if (key0 + lo      > qi) s0 = -__builtin_inff();
        if (key0 + 16 + lo > qi) s1 = -__builtin_inff();

        float rm = fmaxf(s0, s1);
        rm = fmaxf(rm, __shfl_xor(rm, 1));
        rm = fmaxf(rm, __shfl_xor(rm, 2));
        rm = fmaxf(rm, __shfl_xor(rm, 4));
        rm = fmaxf(rm, __shfl_xor(rm, 8));

        float mnew = fmaxf(mrow[g][r], rm);
        float f  = __expf(mrow[g][r] - mnew);
        float p0 = __expf(s0 - mnew);
        float p1 = __expf(s1 - mnew);
        float rs = p0 + p1;
        rs += __shfl_xor(rs, 1);
        rs += __shfl_xor(rs, 2);
        rs += __shfl_xor(rs, 4);
        rs += __shfl_xor(rs, 8);

        lrow[g][r] = lrow[g][r] * f + rs;
        mrow[g][r] = mnew;
#pragma unroll
        for (int td = 0; td < 4; ++td) O[g][td][r] *= f;

        pbuf[g][(r + hi * 8) * 32 + lo]      = (bf16_t)p0;
        pbuf[g][(r + hi * 8) * 32 + 16 + lo] = (bf16_t)p1;
      }
    }

    // Same-wave DS ops are processed in-order; wait before reading back.
    asm volatile("s_wait_dscnt 0" ::: "memory");

    v16bf pa[2];
#pragma unroll
    for (int g = 0; g < 2; ++g) {
      const bf16_t* pp = &pbuf[g][lo * 32 + hi * 16];
      v8bf pl0 = *(const v8bf*)pp;
      v8bf pl1 = *(const v8bf*)(pp + 8);
      pa[g] = __builtin_shufflevector(pl0, pl1, 0,1,2,3,4,5,6,7,8,9,10,11,12,13,14,15);
    }

    // O += P(16x32) * V(32keys x 64d); V fragments shared by both sub-tiles
#pragma unroll
    for (int td = 0; td < 4; ++td) {
      const bf16_t* vp = vt + ((size_t)((b * H_NUM + h) * HD_DIM) + td * 16 + lo) * S_LEN
                            + key0 + hi * 16;
      v16bf vf = *(const v16bf*)vp;
      O[0][td] = wmma_bf(pa[0], vf, O[0][td]);
      O[1][td] = wmma_bf(pa[1], vf, O[1][td]);
    }
  }

  // normalize and store ctx (token-major, merged heads) as bf16
#pragma unroll
  for (int g = 0; g < 2; ++g)
#pragma unroll
    for (int r = 0; r < 8; ++r) {
      float inv = 1.0f / lrow[g][r];
      const int m = q0 + g * 16 + r + hi * 8;
#pragma unroll
      for (int td = 0; td < 4; ++td) {
        ctx[((size_t)(b * S_LEN) + m) * D_DIM + h * HD_DIM + td * 16 + lo] =
            (bf16_t)(O[g][td][r] * inv);
      }
    }
}

// ---------------------------------------------------------------------------
// Output projection: out = ctx * Wo + bo  (fp32 out), 32x64 tile per wave
// ---------------------------------------------------------------------------
__global__ void out_gemm_kernel(const bf16_t* __restrict__ ctx,
                                const bf16_t* __restrict__ wto,
                                const float* __restrict__ bo,
                                float* __restrict__ out) {
  const int lane = threadIdx.x & 31;
  const int wave = threadIdx.x >> 5;
  const int lo = lane & 15, hi = lane >> 4;
  const int m0 = blockIdx.x * 32;
  const int n0 = (blockIdx.y * 8 + wave) * 64;

  v8f acc[2][4] = {};
  for (int k0 = 0; k0 < D_DIM; k0 += 32) {
    v16bf a0 = load_a16x32(ctx, D_DIM, m0,      k0, lo, hi);
    v16bf a1 = load_a16x32(ctx, D_DIM, m0 + 16, k0, lo, hi);
#pragma unroll
    for (int t = 0; t < 4; ++t) {
      v16bf bf = load_bT32x16(wto, D_DIM, n0 + t * 16, k0, lo, hi);
      acc[0][t] = wmma_bf(a0, bf, acc[0][t]);
      acc[1][t] = wmma_bf(a1, bf, acc[1][t]);
    }
  }
#pragma unroll
  for (int t = 0; t < 4; ++t) {
    const int n = n0 + t * 16 + lo;
    const float bv = bo[n];
#pragma unroll
    for (int i = 0; i < 2; ++i)
#pragma unroll
      for (int r = 0; r < 8; ++r) {
        const int m = m0 + i * 16 + r + hi * 8;
        out[(size_t)m * D_DIM + n] = acc[i][t][r] + bv;
      }
  }
}

// ---------------------------------------------------------------------------
// Host launch
// ---------------------------------------------------------------------------
extern "C" void kernel_launch(void* const* d_in, const int* in_sizes, int n_in,
                              void* d_out, int out_size, void* d_ws, size_t ws_size,
                              hipStream_t stream) {
  const float* x  = (const float*)d_in[0];
  const float* Wq = (const float*)d_in[1];
  const float* Wk = (const float*)d_in[2];
  const float* Wv = (const float*)d_in[3];
  const float* Wo = (const float*)d_in[4];
  const float* bo = (const float*)d_in[5];
  float* out = (float*)d_out;

  char* ws = (char*)d_ws;
  const size_t MB = 1024u * 1024u;
  bf16_t* xb  = (bf16_t*)(ws + 0 * MB);    // 8 MiB: x bf16 [4096,1024]
  bf16_t* wtq = (bf16_t*)(ws + 8 * MB);    // 2 MiB: Wq^T bf16 [N][K]
  bf16_t* wtk = (bf16_t*)(ws + 10 * MB);
  bf16_t* wtv = (bf16_t*)(ws + 12 * MB);
  bf16_t* wto = (bf16_t*)(ws + 14 * MB);
  bf16_t* qb  = (bf16_t*)(ws + 16 * MB);   // 8 MiB: Q bf16 [B,S,D]
  bf16_t* kb  = (bf16_t*)(ws + 24 * MB);   // 8 MiB: K bf16 [B,S,D]
  bf16_t* vt  = (bf16_t*)(ws + 32 * MB);   // 8 MiB: V^T bf16 [B,H,HD,S]
  bf16_t* ctx = (bf16_t*)(ws + 40 * MB);   // 8 MiB: ctx bf16 [B,S,D]

  cvt_x_kernel<<<4096, 256, 0, stream>>>(x, xb, BS_TOK * D_DIM);
  transpose_w_kernel<<<4096, 256, 0, stream>>>(Wq, wtq);
  transpose_w_kernel<<<4096, 256, 0, stream>>>(Wk, wtk);
  transpose_w_kernel<<<4096, 256, 0, stream>>>(Wv, wtv);
  transpose_w_kernel<<<4096, 256, 0, stream>>>(Wo, wto);

  qkv_gemm_kernel<<<dim3(BS_TOK / 32, 2, 3), 256, 0, stream>>>(xb, wtq, wtk, wtv, qb, kb, vt);
  attn_kernel<<<dim3(S_LEN / 32, H_NUM, B_NUM), 32, 0, stream>>>(qb, kb, vt, ctx);
  out_gemm_kernel<<<dim3(BS_TOK / 32, 2), 256, 0, stream>>>(ctx, wto, bo, out);
}
